// T5LayerSelfAttention_31559419691200
// MI455X (gfx1250) — compile-verified, matrix operand order checked by reading, not run on hardware
//
#include <hip/hip_runtime.h>
#include <hip/hip_bf16.h>

typedef __attribute__((ext_vector_type(16))) _Float16 v16h;
typedef __attribute__((ext_vector_type(8)))  _Float16 v8h;
typedef __attribute__((ext_vector_type(8)))  float    v8f;

#define NB 4
#define NS 2048
#define ND 1024
#define NH 16
#define NDK 64
#define N_QKV 3072
#define NEG_BIG (-1.0e30f)

static __device__ __forceinline__ v8f wmma_f16(v16h a, v16h b, v8f c) {
  // D = A(16x32 f16) * B(32x16 f16) + C(16x16 f32)
  return __builtin_amdgcn_wmma_f32_16x16x32_f16(false, a, false, b, (short)0, c,
                                                false, false);
}

// Load a 16x32 f16 A-fragment (or, for N-major storage, a 32x16 B-fragment)
// from row-major memory with leading dimension `ld` (multiple of 8).
// Layout per CDNA5 ISA 7.12.2: lane = {m = lane&15, half = lane>>4};
// elements e<8 -> K = 8*half + e ; e>=8 -> K = 16 + 8*half + (e-8).
static __device__ __forceinline__ v16h load_frag(const _Float16* base, int ld, int lane) {
  const int m  = lane & 15;
  const int hh = lane >> 4;
  const _Float16* p = base + (size_t)m * ld + hh * 8;
  v8h lo = *(const v8h*)(p);
  v8h hi = *(const v8h*)(p + 16);
  v16h r;
#pragma unroll
  for (int i = 0; i < 8; ++i) { r[i] = lo[i]; r[i + 8] = hi[i]; }
  return r;
}

// ---------------- RMSNorm -> f16 ----------------
__global__ __launch_bounds__(256) void rmsnorm_f16_kernel(const float* __restrict__ hs,
                                                          const float* __restrict__ w,
                                                          _Float16* __restrict__ out) {
  const int row = blockIdx.x;
  const float* x = hs + (size_t)row * ND;
  float ss = 0.f;
  for (int d = threadIdx.x; d < ND; d += 256) { float v = x[d]; ss += v * v; }
  __shared__ float red[256];
  red[threadIdx.x] = ss;
  __syncthreads();
  for (int s = 128; s > 0; s >>= 1) {
    if (threadIdx.x < s) red[threadIdx.x] += red[threadIdx.x + s];
    __syncthreads();
  }
  const float inv = rsqrtf(red[0] * (1.0f / ND) + 1e-6f);
  _Float16* o = out + (size_t)row * ND;
  for (int d = threadIdx.x; d < ND; d += 256)
    o[d] = (_Float16)(x[d] * inv * w[d]);
}

// ------------- W[k][n] f32 -> Wt[n][k] f16 -------------
__global__ void transpose_to_f16_kernel(const float* __restrict__ W,
                                        _Float16* __restrict__ Wt, int K, int N) {
  int i = blockIdx.x * 256 + threadIdx.x;
  if (i >= K * N) return;
  int k = i / N, n = i - k * N;
  Wt[(size_t)n * K + k] = (_Float16)W[i];
}

// ------------- T5 relative bias table: tab[h][d], d = q - k >= 0 -------------
__global__ void bias_table_kernel(const float* __restrict__ rel_bias,
                                  float* __restrict__ tab) {
  int i = blockIdx.x * 256 + threadIdx.x;
  if (i >= NH * NS) return;
  int h = i / NS, d = i - h * NS;
  int bucket;
  if (d < 16) {
    bucket = d;
  } else {
    // 16 + log(d/16)/log(128/16) * 16, truncated, clamped to 31
    float v = 16.0f + (logf((float)d * (1.0f / 16.0f)) * (1.0f / logf(8.0f))) * 16.0f;
    bucket = (int)v;
    if (bucket > 31) bucket = 31;
  }
  tab[(size_t)h * NS + d] = rel_bias[bucket * NH + h];
}

// ------------- QKV GEMM: [8192 x 1024] x [1024 x 3072], 64x64 tile/wave -------------
__global__ __launch_bounds__(256) void gemm_qkv_kernel(const _Float16* __restrict__ X,
                                                       const _Float16* __restrict__ Wt,
                                                       _Float16* __restrict__ Qo,
                                                       _Float16* __restrict__ Ko,
                                                       _Float16* __restrict__ Vt) {
  const int lane = threadIdx.x & 31;
  const int wave = threadIdx.x >> 5;
  const int M = NB * NS, K = ND, N = N_QKV;
  const int mTiles = M / 64, nTiles = N / 64;
  int tile = blockIdx.x * 8 + wave;
  if (tile >= mTiles * nTiles) return;
  const int mt = tile % mTiles, nt = tile / mTiles;   // consecutive waves share B panel
  const _Float16* Ab = X + (size_t)(mt * 64) * K;
  const _Float16* Bb = Wt + (size_t)(nt * 64) * K;
  v8f acc[4][4] = {};
  for (int k = 0; k < K; k += 32) {
    v16h a[4], b[4];
#pragma unroll
    for (int i = 0; i < 4; ++i) a[i] = load_frag(Ab + (size_t)(16 * i) * K + k, K, lane);
#pragma unroll
    for (int j = 0; j < 4; ++j) b[j] = load_frag(Bb + (size_t)(16 * j) * K + k, K, lane);
#pragma unroll
    for (int i = 0; i < 4; ++i)
#pragma unroll
      for (int j = 0; j < 4; ++j)
        acc[i][j] = wmma_f16(a[i], b[j], acc[i][j]);
  }
  const int hh = lane >> 4, cn = lane & 15;
#pragma unroll
  for (int i = 0; i < 4; ++i)
#pragma unroll
    for (int j = 0; j < 4; ++j)
#pragma unroll
      for (int v = 0; v < 8; ++v) {
        const int m = mt * 64 + i * 16 + v + 8 * hh;
        const int n = nt * 64 + j * 16 + cn;
        const int b2 = m >> 11, s = m & (NS - 1);
        const int which = n >> 10, rem = n & 1023;
        const int h = rem >> 6, dk = rem & 63;
        const _Float16 val = (_Float16)acc[i][j][v];
        const size_t bh = (size_t)(b2 * NH + h);
        if (which == 0)      Qo[(bh * NS + s) * NDK + dk] = val;  // [b,h,s,dk]
        else if (which == 1) Ko[(bh * NS + s) * NDK + dk] = val;  // [b,h,s,dk]
        else                 Vt[(bh * NDK + dk) * NS + s] = val;  // [b,h,dk,s]
      }
}

// ------------- Flash attention: 1 wave per 32-query tile -------------
__global__ __launch_bounds__(256) void attn_kernel(const _Float16* __restrict__ Q,
                                                   const _Float16* __restrict__ Km,
                                                   const _Float16* __restrict__ Vt,
                                                   const float* __restrict__ biasTab,
                                                   const unsigned char* __restrict__ seqmask,
                                                   _Float16* __restrict__ ctx) {
  __shared__ _Float16 plds[8][1024];  // per wave: two 16x32 f16 P tiles, row-major
  const int lane = threadIdx.x & 31;
  const int wave = threadIdx.x >> 5;
  const int tile = blockIdx.x * 8 + wave;     // [0, B*H*(S/32))
  const int qt = tile & (NS / 32 - 1);        // S/32 = 64
  const int bh = tile >> 6;
  const int b = bh >> 4, h = bh & 15;
  const int qBase = qt * 32;
  const _Float16* Qb = Q  + (size_t)bh * NS * NDK;
  const _Float16* Kb = Km + (size_t)bh * NS * NDK;
  const _Float16* Vb = Vt + (size_t)bh * NDK * NS;
  const float* bt = biasTab + (size_t)h * NS;
  const unsigned char* msk = seqmask + (size_t)b * NS;
  const int hh = lane >> 4, cn = lane & 15;

  v16h qf[2][2];  // [qsub][dk-half]
#pragma unroll
  for (int u = 0; u < 2; ++u) {
    qf[u][0] = load_frag(Qb + (size_t)(qBase + 16 * u) * NDK, NDK, lane);
    qf[u][1] = load_frag(Qb + (size_t)(qBase + 16 * u) * NDK + 32, NDK, lane);
  }

  float mrow[2][8], lrow[2][8];
#pragma unroll
  for (int u = 0; u < 2; ++u)
#pragma unroll
    for (int v = 0; v < 8; ++v) { mrow[u][v] = NEG_BIG; lrow[u][v] = 0.f; }
  v8f acc[2][4] = {};  // [qsub][dk tile]

  const int kEnd = qBase + 32;          // causal: keys 0..qBase+31
  for (int kb = 0; kb < kEnd; kb += 32) {
    // ---- scores: 2 q-subtiles x 2 key-subtiles ----
    v8f s[2][2];
    {
      v16h kf[2][2];  // [ksub][dk-half]
#pragma unroll
      for (int t = 0; t < 2; ++t) {
        kf[t][0] = load_frag(Kb + (size_t)(kb + 16 * t) * NDK, NDK, lane);
        kf[t][1] = load_frag(Kb + (size_t)(kb + 16 * t) * NDK + 32, NDK, lane);
      }
#pragma unroll
      for (int u = 0; u < 2; ++u)
#pragma unroll
        for (int t = 0; t < 2; ++t) {
          v8f z = {};
          z = wmma_f16(qf[u][0], kf[t][0], z);
          z = wmma_f16(qf[u][1], kf[t][1], z);
          s[u][t] = z;
        }
    }
    const int c0 = kb + cn, c1 = kb + 16 + cn;
    const float mk0 = msk[c0] ? 0.f : NEG_BIG;
    const float mk1 = msk[c1] ? 0.f : NEG_BIG;
    // ---- per q-subtile: bias/mask, online softmax, rescale, P -> LDS ----
#pragma unroll
    for (int u = 0; u < 2; ++u) {
      float p0[8], p1[8];
#pragma unroll
      for (int v = 0; v < 8; ++v) {
        const int qrow = qBase + 16 * u + v + 8 * hh;
        const int d0 = qrow - c0, d1 = qrow - c1;
        p0[v] = (d0 >= 0) ? (s[u][0][v] + bt[d0] + mk0) : NEG_BIG;
        p1[v] = (d1 >= 0) ? (s[u][1][v] + bt[d1] + mk1) : NEG_BIG;
      }
      float scale[8];
#pragma unroll
      for (int v = 0; v < 8; ++v) {
        float tm = fmaxf(p0[v], p1[v]);
#pragma unroll
        for (int off = 8; off >= 1; off >>= 1)
          tm = fmaxf(tm, __shfl_xor(tm, off));
        const float mnew = fmaxf(mrow[u][v], tm);
        const float sc = __expf(mrow[u][v] - mnew);
        const float e0 = __expf(p0[v] - mnew);
        const float e1 = __expf(p1[v] - mnew);
        float rs = e0 + e1;
#pragma unroll
        for (int off = 8; off >= 1; off >>= 1)
          rs += __shfl_xor(rs, off);
        lrow[u][v] = lrow[u][v] * sc + rs;
        mrow[u][v] = mnew;
        scale[v] = sc;
        p0[v] = e0;
        p1[v] = e1;
      }
#pragma unroll
      for (int t = 0; t < 4; ++t)
#pragma unroll
        for (int v = 0; v < 8; ++v) acc[u][t][v] *= scale[v];
      _Float16* my = &plds[wave][u * 512];
#pragma unroll
      for (int v = 0; v < 8; ++v) {
        const int r = v + 8 * hh;
        my[r * 32 + cn]      = (_Float16)p0[v];
        my[r * 32 + 16 + cn] = (_Float16)p1[v];
      }
    }
    asm volatile("" ::: "memory");  // LDS per-wave in-order; block compiler reordering
    v16h pf[2];
    pf[0] = load_frag(&plds[wave][0], 32, lane);
    pf[1] = load_frag(&plds[wave][512], 32, lane);
    // ---- ctx += P * V ; each V fragment feeds both q-subtiles ----
#pragma unroll
    for (int t = 0; t < 4; ++t) {
      v16h vf = load_frag(Vb + (size_t)(t * 16) * NS + kb, NS, lane);
      acc[0][t] = wmma_f16(pf[0], vf, acc[0][t]);
      acc[1][t] = wmma_f16(pf[1], vf, acc[1][t]);
    }
    asm volatile("" ::: "memory");
  }
  // ---- normalize and store ctx f16, token-major [b,s,h*64+dk] ----
#pragma unroll
  for (int u = 0; u < 2; ++u)
#pragma unroll
    for (int v = 0; v < 8; ++v) {
      const int s = qBase + 16 * u + v + 8 * hh;
      const float invl = 1.0f / lrow[u][v];
      _Float16* orow = ctx + (size_t)(b * NS + s) * (NH * NDK) + h * NDK;
#pragma unroll
      for (int t = 0; t < 4; ++t)
        orow[t * 16 + cn] = (_Float16)(acc[u][t][v] * invl);
    }
}

// ------------- Output GEMM + residual: out = hidden + ctx @ Wo, 64x64 tile/wave -------------
__global__ __launch_bounds__(256) void gemm_out_kernel(const _Float16* __restrict__ X,
                                                       const _Float16* __restrict__ Wt,
                                                       const float* __restrict__ resid,
                                                       float* __restrict__ out) {
  const int lane = threadIdx.x & 31;
  const int wave = threadIdx.x >> 5;
  const int M = NB * NS, K = ND, N = ND;
  const int mTiles = M / 64, nTiles = N / 64;
  int tile = blockIdx.x * 8 + wave;
  if (tile >= mTiles * nTiles) return;
  const int mt = tile % mTiles, nt = tile / mTiles;
  const _Float16* Ab = X + (size_t)(mt * 64) * K;
  const _Float16* Bb = Wt + (size_t)(nt * 64) * K;
  v8f acc[4][4] = {};
  for (int k = 0; k < K; k += 32) {
    v16h a[4], b[4];
#pragma unroll
    for (int i = 0; i < 4; ++i) a[i] = load_frag(Ab + (size_t)(16 * i) * K + k, K, lane);
#pragma unroll
    for (int j = 0; j < 4; ++j) b[j] = load_frag(Bb + (size_t)(16 * j) * K + k, K, lane);
#pragma unroll
    for (int i = 0; i < 4; ++i)
#pragma unroll
      for (int j = 0; j < 4; ++j)
        acc[i][j] = wmma_f16(a[i], b[j], acc[i][j]);
  }
  const int hh = lane >> 4, cn = lane & 15;
#pragma unroll
  for (int i = 0; i < 4; ++i)
#pragma unroll
    for (int j = 0; j < 4; ++j)
#pragma unroll
      for (int v = 0; v < 8; ++v) {
        const int m = mt * 64 + i * 16 + v + 8 * hh;
        const int n = nt * 64 + j * 16 + cn;
        out[(size_t)m * N + n] = resid[(size_t)m * N + n] + acc[i][j][v];
      }
}

extern "C" void kernel_launch(void* const* d_in, const int* in_sizes, int n_in,
                              void* d_out, int out_size, void* d_ws, size_t ws_size,
                              hipStream_t stream) {
  const float* hs           = (const float*)d_in[0];
  const unsigned char* mask = (const unsigned char*)d_in[1];
  const float* rmsw         = (const float*)d_in[2];
  const float* Wqkv         = (const float*)d_in[3];
  const float* Wo           = (const float*)d_in[4];
  const float* relb         = (const float*)d_in[5];
  float* out = (float*)d_out;

  char* p = (char*)d_ws;
  auto take = [&](size_t bytes) {
    char* r = p;
    p += (bytes + 255) & ~(size_t)255;
    return r;
  };
  _Float16* normed = (_Float16*)take((size_t)NB * NS * ND * 2);
  _Float16* WqkvT  = (_Float16*)take((size_t)ND * N_QKV * 2);
  _Float16* WoT    = (_Float16*)take((size_t)ND * ND * 2);
  _Float16* Qf     = (_Float16*)take((size_t)NB * NH * NS * NDK * 2);
  _Float16* Kf     = (_Float16*)take((size_t)NB * NH * NS * NDK * 2);
  _Float16* Vtf    = (_Float16*)take((size_t)NB * NH * NDK * NS * 2);
  float*    btab   = (float*)take((size_t)NH * NS * 4);
  _Float16* ctxf   = (_Float16*)take((size_t)NB * NS * ND * 2);

  rmsnorm_f16_kernel<<<NB * NS, 256, 0, stream>>>(hs, rmsw, normed);
  transpose_to_f16_kernel<<<(ND * N_QKV + 255) / 256, 256, 0, stream>>>(Wqkv, WqkvT, ND, N_QKV);
  transpose_to_f16_kernel<<<(ND * ND + 255) / 256, 256, 0, stream>>>(Wo, WoT, ND, ND);
  bias_table_kernel<<<(NH * NS + 255) / 256, 256, 0, stream>>>(relb, btab);

  {
    const int tiles = (NB * NS / 64) * (N_QKV / 64);
    gemm_qkv_kernel<<<(tiles + 7) / 8, 256, 0, stream>>>(normed, WqkvT, Qf, Kf, Vtf);
  }
  attn_kernel<<<NB * NH * (NS / 32) / 8, 256, 0, stream>>>(Qf, Kf, Vtf, btab, mask, ctxf);
  {
    const int tiles = (NB * NS / 64) * (ND / 64);
    gemm_out_kernel<<<(tiles + 7) / 8, 256, 0, stream>>>(ctxf, WoT, hs, out);
  }
}
